// MaskedTemporalSelfAttention_78675210928335
// MI455X (gfx1250) — matrix operand order repaired and checked by reading, not compiled, not run on hardware
//
#include <hip/hip_runtime.h>

// ---------------- problem constants (from reference) ----------------
#define T_   16
#define N_   20000
#define E_   320000
#define P_   4096
#define DIN_ 128
#define H_   256
#define NH_  8
#define L_   2
#define FF_  2048
#define D_   512      // 2*H
#define HD_  64       // D/NH

#define GF_BIAS 1
#define GF_RELU 2
#define GF_ACC  4

typedef __attribute__((ext_vector_type(16))) _Float16 v16h;
typedef __attribute__((ext_vector_type(8)))  _Float16 v8h;
typedef __attribute__((ext_vector_type(4)))  _Float16 v4h;
typedef __attribute__((ext_vector_type(8)))  float    v8f;

union AFrag { v16h v; v8h h[2]; _Float16 e[16]; };
union CFrag { v8f  v; float e[8]; };

// A-fragment k-index for V_WMMA_*_16X16X32_F16 (16-bit A 16x32 layout, ISA 7.12.2):
// lanes 0-15: v0..v3 -> K 0..7, v4..v7 -> K 16..23 ; lanes 16-31: +8 within group.
__device__ __forceinline__ int a_frag_k(int j, int half) {
  int v = j >> 1, pos = j & 1;
  return ((v < 4) ? 0 : 16) + half * 8 + ((v & 3) << 1) + pos;
}

// ---------------- generic WMMA GEMM: C[MxN] (= or +=) A[MxK] @ B[KxN] ----------------
// block = 256 threads (8 waves). Block tile 128x128, wave tile 32x64 (2x4 WMMA frags).
// REQUIRES: M % 128 == 0, N % 128 == 0, K % 32 == 0 (true for every GEMM here).
// f32 inputs converted to f16 while staging LDS; f32 accumulate.
__global__ __launch_bounds__(256) void k_gemm(const float* __restrict__ A,
                                              const float* __restrict__ B,
                                              const float* __restrict__ bias,
                                              float* __restrict__ C,
                                              int M, int Nn, int K, int flags) {
  (void)M;
  // A tile: [m][k] row-padded to 40 halves (80B, 16B-aligned rows).
  __shared__ __align__(16) _Float16 As[128][40];
  // B tile pre-swizzled in per-lane B-fragment order (contiguous k, as the A loads):
  // BsX[nChunk][lane][j]  with lane = (k>>4)*16 + (n&15), j = k&15.
  // Row stride 24 halves (48B) keeps every 16-half fragment 16B-aligned.
  __shared__ __align__(16) _Float16 BsX[8][32][24];

  const int tid  = threadIdx.x;
  const int lane = tid & 31, wv = tid >> 5;
  const int wr = wv >> 1, wc = wv & 1;           // 4x2 wave grid over 128x128
  const int half = lane >> 4, l15 = lane & 15;
  const long long bm = (long long)blockIdx.y * 128;
  const long long bn = (long long)blockIdx.x * 128;

  CFrag acc[2][4];
#pragma unroll
  for (int i = 0; i < 2; ++i)
#pragma unroll
    for (int j = 0; j < 4; ++j)
#pragma unroll
      for (int v = 0; v < 8; ++v) acc[i][j].e[v] = 0.f;

  for (int k0 = 0; k0 < K; k0 += 32) {
    if (k0 + 32 < K) {                           // gfx1250 global_prefetch path
      __builtin_prefetch(&A[(bm + (tid >> 1)) * (long long)K + (k0 + 32) + (tid & 1) * 16], 0, 1);
      __builtin_prefetch(&B[(long long)(k0 + 32 + (tid >> 5)) * Nn + bn + (tid & 31) * 4], 0, 1);
    }
    // ---- A tile: 128x32 as 1024 float4 chunks (4 per thread) ----
#pragma unroll
    for (int it = 0; it < 4; ++it) {
      int i = tid + it * 256;
      int m = i >> 3, c = i & 7;
      const float4 a4 = *(const float4*)&A[(bm + m) * (long long)K + k0 + c * 4];
      v4h hv = {(_Float16)a4.x, (_Float16)a4.y, (_Float16)a4.z, (_Float16)a4.w};
      *(v4h*)&As[m][c * 4] = hv;
    }
    // ---- B tile: 32x128, 4 consecutive k per thread for one column n ----
#pragma unroll
    for (int it = 0; it < 4; ++it) {
      int i = tid + it * 256;                    // 0..1023
      int n = i & 127;
      int kg = i >> 7;                           // 0..7 -> k = 4*kg..4*kg+3 (same k-half)
      int k = kg * 4;
      long long col = bn + n;
      float b0 = B[(long long)(k0 + k + 0) * Nn + col];
      float b1 = B[(long long)(k0 + k + 1) * Nn + col];
      float b2 = B[(long long)(k0 + k + 2) * Nn + col];
      float b3 = B[(long long)(k0 + k + 3) * Nn + col];
      v4h hv = {(_Float16)b0, (_Float16)b1, (_Float16)b2, (_Float16)b3};
      *(v4h*)&BsX[n >> 4][(k >> 4) * 16 + (n & 15)][k & 15] = hv;
    }
    __syncthreads();

    AFrag af[2], bf[4];
#pragma unroll
    for (int i = 0; i < 2; ++i) {                // A frags: 2x ds_load_b128 each
      int m = wr * 32 + i * 16 + l15;
      af[i].h[0] = *(const v8h*)&As[m][half * 16];
      af[i].h[1] = *(const v8h*)&As[m][half * 16 + 8];
    }
#pragma unroll
    for (int j = 0; j < 4; ++j) {                // B frags: 2x ds_load_b128 each
      int nb = wc * 4 + j;
      bf[j].h[0] = *(const v8h*)&BsX[nb][lane][0];
      bf[j].h[1] = *(const v8h*)&BsX[nb][lane][8];
    }
#pragma unroll
    for (int i = 0; i < 2; ++i)
#pragma unroll
      for (int j = 0; j < 4; ++j)
        acc[i][j].v = __builtin_amdgcn_wmma_f32_16x16x32_f16(
            false, af[i].v, false, bf[j].v, (short)0, acc[i][j].v, false, false);
    __syncthreads();
  }

  // epilogue (exact tiling: no bounds checks)
#pragma unroll
  for (int i = 0; i < 2; ++i)
#pragma unroll
    for (int j = 0; j < 4; ++j) {
      long long gn = bn + wc * 64 + j * 16 + l15;
      float bval = (flags & GF_BIAS) ? bias[gn] : 0.f;
#pragma unroll
      for (int v = 0; v < 8; ++v) {
        long long gm = bm + wr * 32 + i * 16 + v + 8 * half;
        long long idx = gm * Nn + gn;
        float val = acc[i][j].e[v] + bval;
        if (flags & GF_ACC)  val += C[idx];
        if (flags & GF_RELU) val = fmaxf(val, 0.f);
        C[idx] = val;
      }
    }
}

// ---------------- banded attention, one wave per (p, head) ----------------
// S = K·Qᵀ (two 16x16x32 WMMAs over HD=64) -> D[s][t]; mask+softmax over s in the
// C-fragment (lane pair l, l^16 share column t); probs staged via LDS into an
// A-fragment (K padded 16->32 with zeros); O = P·V via 4 WMMAs (d in chunks of 16).
__global__ __launch_bounds__(256) void k_attn(const float* __restrict__ Q,
                                              const float* __restrict__ Km,
                                              const float* __restrict__ Vm,
                                              float* __restrict__ O) {
  __shared__ _Float16 Pm[8][16][18];
  const int tid = threadIdx.x;
  const int lane = tid & 31, wv = tid >> 5;
  const int half = lane >> 4, l15 = lane & 15;
  const int gw = blockIdx.x * 8 + wv;            // p*NH + head
  const int p = gw >> 3, head = gw & 7;
  const long long rowStride = (long long)P_ * D_;
  const long long base = (long long)p * D_ + head * HD_;

  CFrag sc;
#pragma unroll
  for (int v = 0; v < 8; ++v) sc.e[v] = 0.f;

#pragma unroll
  for (int kk = 0; kk < 2; ++kk) {
    AFrag ka, qb;
    // A = K rows (m = s = l15): 8 consecutive-d pairs -> float2 loads
#pragma unroll
    for (int v = 0; v < 8; ++v) {
      int d = kk * 32 + half * 16 + (v << 1);
      const float2 kv = *(const float2*)&Km[(long long)l15 * rowStride + base + d];
      ka.e[2 * v]     = (_Float16)kv.x;
      ka.e[2 * v + 1] = (_Float16)kv.y;
    }
    // B = Qᵀ cols (n = t = l15): 16 consecutive d -> 4x float4 loads
    const float4* qp = (const float4*)&Q[(long long)l15 * rowStride + base + kk * 32 + half * 16];
#pragma unroll
    for (int c = 0; c < 4; ++c) {
      const float4 q4 = qp[c];
      qb.e[c * 4 + 0] = (_Float16)q4.x;
      qb.e[c * 4 + 1] = (_Float16)q4.y;
      qb.e[c * 4 + 2] = (_Float16)q4.z;
      qb.e[c * 4 + 3] = (_Float16)q4.w;
    }
    sc.v = __builtin_amdgcn_wmma_f32_16x16x32_f16(false, ka.v, false, qb.v,
                                                  (short)0, sc.v, false, false);
  }

  // sc holds S[s][t]: s = v + 8*half, t = l15. Mask: allowed iff 0 <= t-s < 4.
  const float scale = 0.125f;                    // 1/sqrt(64)
  float mx = -3.0e38f;
#pragma unroll
  for (int v = 0; v < 8; ++v) {
    int s = v + 8 * half, t = l15;
    bool ok = (s <= t) && (t - s < 4);
    float e = ok ? sc.e[v] * scale : -3.0e38f;
    sc.e[v] = e;
    mx = fmaxf(mx, e);
  }
  mx = fmaxf(mx, __shfl_xor(mx, 16));
  float sum = 0.f;
#pragma unroll
  for (int v = 0; v < 8; ++v) {
    float e = __expf(sc.e[v] - mx);
    sc.e[v] = e;
    sum += e;
  }
  sum += __shfl_xor(sum, 16);
  const float inv = 1.f / sum;
#pragma unroll
  for (int v = 0; v < 8; ++v)
    Pm[wv][l15][v + 8 * half] = (_Float16)(sc.e[v] * inv);
  __syncthreads();

  AFrag pa;                                      // A = P (m = t, k = s, pad k>=16 -> 0)
#pragma unroll
  for (int j = 0; j < 16; ++j) {
    int kq = half * 16 + j;
    pa.e[j] = (kq < 16) ? Pm[wv][l15][kq] : (_Float16)0.f;
  }
#pragma unroll
  for (int ch = 0; ch < 4; ++ch) {
    AFrag vbf;                                   // B = V (k = s, n = d in 16-chunk)
#pragma unroll
    for (int j = 0; j < 16; ++j)
      vbf.e[j] = (half == 0)
          ? (_Float16)Vm[(long long)j * rowStride + base + ch * 16 + l15]
          : (_Float16)0.f;
    CFrag o;
#pragma unroll
    for (int v = 0; v < 8; ++v) o.e[v] = 0.f;
    o.v = __builtin_amdgcn_wmma_f32_16x16x32_f16(false, pa.v, false, vbf.v,
                                                 (short)0, o.v, false, false);
#pragma unroll
    for (int v = 0; v < 8; ++v) {
      int t = v + 8 * half;
      O[(long long)t * rowStride + base + ch * 16 + l15] = o.e[v];
    }
  }
}

// ---------------- graph aggregation helpers ----------------
__global__ void k_deg(const int* __restrict__ dst, float* __restrict__ deg) {
  long long gid = (long long)blockIdx.x * blockDim.x + threadIdx.x;
  if (gid >= (long long)T_ * E_) return;
  long long t = gid / E_;
  atomicAdd(&deg[t * N_ + dst[gid]], 1.0f);
}

__global__ void k_scatter(const float* __restrict__ x, const int* __restrict__ src,
                          const int* __restrict__ dst, float* __restrict__ agg, int dim) {
  const int ch = dim >> 2;
  long long gid = (long long)blockIdx.x * blockDim.x + threadIdx.x;
  if (gid >= (long long)T_ * E_ * ch) return;
  int c = (int)(gid % ch);
  long long te = gid / ch;
  long long t = te / E_;
  int s = src[te], d = dst[te];
  const float4 val = *(const float4*)&x[((t * N_ + s) * (long long)dim) + c * 4];
  float* o = &agg[((t * N_ + d) * (long long)dim) + c * 4];
  atomicAdd(o + 0, val.x);
  atomicAdd(o + 1, val.y);
  atomicAdd(o + 2, val.z);
  atomicAdd(o + 3, val.w);
}

__global__ void k_scale(float* __restrict__ agg, const float* __restrict__ deg,
                        long long rows, int dim) {
  long long gid = (long long)blockIdx.x * blockDim.x + threadIdx.x;
  if (gid >= rows * dim) return;
  long long r = gid / dim;
  agg[gid] *= 1.f / fmaxf(deg[r], 1.f);
}

// ---------------- row-wise kernels (one wave per row) ----------------
__global__ __launch_bounds__(256) void k_l2norm(float* __restrict__ h, long long rows) {
  int lane = threadIdx.x & 31, wv = threadIdx.x >> 5;
  long long row = (long long)blockIdx.x * 8 + wv;
  if (row >= rows) return;
  long long base = row * H_;
  float x[8], ss = 0.f;
#pragma unroll
  for (int i = 0; i < 8; ++i) { x[i] = h[base + i * 32 + lane]; ss += x[i] * x[i]; }
#pragma unroll
  for (int off = 16; off; off >>= 1) ss += __shfl_xor(ss, off);
  float inv = 1.f / fmaxf(sqrtf(ss), 1e-12f);
#pragma unroll
  for (int i = 0; i < 8; ++i) h[base + i * 32 + lane] = x[i] * inv;
}

__global__ __launch_bounds__(256) void k_ln_res(float* __restrict__ h, const float* __restrict__ r,
                                                const float* __restrict__ g, const float* __restrict__ b) {
  int lane = threadIdx.x & 31, wv = threadIdx.x >> 5;
  long long row = (long long)blockIdx.x * 8 + wv;
  if (row >= (long long)T_ * P_) return;
  long long base = row * D_;
  float x[16], s = 0.f;
#pragma unroll
  for (int i = 0; i < 16; ++i) {
    int d = i * 32 + lane;
    x[i] = h[base + d] + r[base + d];
    s += x[i];
  }
#pragma unroll
  for (int off = 16; off; off >>= 1) s += __shfl_xor(s, off);
  float mean = s * (1.f / D_);
  float vs = 0.f;
#pragma unroll
  for (int i = 0; i < 16; ++i) { float dd = x[i] - mean; vs += dd * dd; }
#pragma unroll
  for (int off = 16; off; off >>= 1) vs += __shfl_xor(vs, off);
  float rstd = rsqrtf(vs * (1.f / D_) + 1e-5f);
#pragma unroll
  for (int i = 0; i < 16; ++i) {
    int d = i * 32 + lane;
    h[base + d] = (x[i] - mean) * rstd * g[d] + b[d];
  }
}

__global__ __launch_bounds__(256) void k_rowdot(const float* __restrict__ h,
                                                const float* __restrict__ w,
                                                float* __restrict__ o) {
  int lane = threadIdx.x & 31, wv = threadIdx.x >> 5;
  long long row = (long long)blockIdx.x * 8 + wv;
  if (row >= (long long)T_ * P_) return;
  float s = 0.f;
#pragma unroll
  for (int i = 0; i < 16; ++i) s += h[row * D_ + i * 32 + lane] * w[i * 32 + lane];
#pragma unroll
  for (int off = 16; off; off >>= 1) s += __shfl_xor(s, off);
  if (lane == 0) o[row] = s;
}

// gather normalized SAGE features into att_in (d_out) and the transformer buffer
__global__ void k_gather(const float* __restrict__ h2, const int* __restrict__ idx,
                         float* __restrict__ att, float* __restrict__ hb, int colOff) {
  long long gid = (long long)blockIdx.x * blockDim.x + threadIdx.x;
  if (gid >= (long long)T_ * P_ * (H_ / 4)) return;
  int c = (int)(gid & 63);
  long long tp = gid >> 6;                       // t*P + p
  long long t = tp >> 12;                        // P_ = 4096
  int node = idx[tp];
  const float4 v = *(const float4*)&h2[(t * N_ + node) * (long long)H_ + c * 4];
  long long o = tp * D_ + colOff + c * 4;
  *(float4*)&att[o] = v;
  *(float4*)&hb[o]  = v;
}

// ---------------- host launcher ----------------
extern "C" void kernel_launch(void* const* d_in, const int* in_sizes, int n_in,
                              void* d_out, int out_size, void* d_ws, size_t ws_size,
                              hipStream_t stream) {
  (void)in_sizes; (void)n_in; (void)out_size; (void)ws_size;
  const float* x_i  = (const float*)d_in[0];
  const float* x_j  = (const float*)d_in[1];
  const int* src_i  = (const int*)d_in[2];
  const int* dst_i  = (const int*)d_in[3];
  const int* src_j  = (const int*)d_in[4];
  const int* dst_j  = (const int*)d_in[5];
  const int* idx_i  = (const int*)d_in[6];
  const int* idx_j  = (const int*)d_in[7];
  const float* W1s  = (const float*)d_in[8];
  const float* W1n  = (const float*)d_in[9];
  const float* b1   = (const float*)d_in[10];
  const float* W2s  = (const float*)d_in[11];
  const float* W2n  = (const float*)d_in[12];
  const float* b2   = (const float*)d_in[13];
  const float* Wq   = (const float*)d_in[14];
  const float* Wk   = (const float*)d_in[15];
  const float* Wv   = (const float*)d_in[16];
  const float* bq   = (const float*)d_in[17];
  const float* bk   = (const float*)d_in[18];
  const float* bv   = (const float*)d_in[19];
  const float* Wo   = (const float*)d_in[20];
  const float* bo   = (const float*)d_in[21];
  const float* Wf1  = (const float*)d_in[22];
  const float* bf1  = (const float*)d_in[23];
  const float* Wf2  = (const float*)d_in[24];
  const float* bf2  = (const float*)d_in[25];
  const float* g1   = (const float*)d_in[26];
  const float* bg1  = (const float*)d_in[27];
  const float* g2   = (const float*)d_in[28];
  const float* bg2  = (const float*)d_in[29];
  const float* Wout = (const float*)d_in[30];

  float* out = (float*)d_out;                    // [T*P*D att_in | T*P out]
  float* ws  = (float*)d_ws;

  const long long TN = (long long)T_ * N_;       // 320000
  const long long TP = (long long)T_ * P_;       // 65536

  // SAGE-phase buffers (floats): agg | h1 | h2 | deg , then persistent hb.
  float* agg = ws;
  float* h1  = ws + TN * 256;
  float* h2  = ws + 2 * TN * 256;
  float* deg = ws + 3 * TN * 256;
  float* hb  = ws + 3 * TN * 256 + TN;           // transformer hidden [TP, D]
  // Transformer-phase overlay of the (dead) SAGE region:
  float* qb   = ws;
  float* kb   = ws + 1 * TP * (long long)D_;
  float* vb   = ws + 2 * TP * (long long)D_;
  float* atb  = ws + 3 * TP * (long long)D_;
  float* tmpb = ws + 4 * TP * (long long)D_;
  float* ffb  = ws;                              // FF phase: [TP, FF] over q/k/v/at slots

  dim3 blk(256);
  auto gemm = [&](const float* A, const float* B, const float* bias, float* C,
                  long long M, int Nn, int K, int flags) {
    dim3 grid((unsigned)((Nn + 127) / 128), (unsigned)((M + 127) / 128));
    k_gemm<<<grid, blk, 0, stream>>>(A, B, bias, C, (int)M, Nn, K, flags);
  };

  auto sage = [&](const float* x, const int* src, const int* dst, const int* idx, int colOff) {
    const long long ne = (long long)T_ * E_;
    hipMemsetAsync(agg, 0, TN * 128 * sizeof(float), stream);
    hipMemsetAsync(deg, 0, TN * sizeof(float), stream);
    k_deg<<<dim3((unsigned)((ne + 255) / 256)), blk, 0, stream>>>(dst, deg);
    k_scatter<<<dim3((unsigned)((ne * 32 + 255) / 256)), blk, 0, stream>>>(x, src, dst, agg, 128);
    k_scale<<<dim3((unsigned)((TN * 128 + 255) / 256)), blk, 0, stream>>>(agg, deg, TN, 128);
    gemm(x,   W1s, b1, h1, TN, 256, 128, GF_BIAS);
    gemm(agg, W1n, nullptr, h1, TN, 256, 128, GF_ACC | GF_RELU);
    hipMemsetAsync(agg, 0, TN * 256 * sizeof(float), stream);
    k_scatter<<<dim3((unsigned)((ne * 64 + 255) / 256)), blk, 0, stream>>>(h1, src, dst, agg, 256);
    k_scale<<<dim3((unsigned)((TN * 256 + 255) / 256)), blk, 0, stream>>>(agg, deg, TN, 256);
    gemm(h1,  W2s, b2, h2, TN, 256, 256, GF_BIAS);
    gemm(agg, W2n, nullptr, h2, TN, 256, 256, GF_ACC);
    k_l2norm<<<dim3((unsigned)((TN + 7) / 8)), blk, 0, stream>>>(h2, TN);
    k_gather<<<dim3((unsigned)((TP * 64 + 255) / 256)), blk, 0, stream>>>(h2, idx, out, hb, colOff);
  };

  sage(x_i, src_i, dst_i, idx_i, 0);
  sage(x_j, src_j, dst_j, idx_j, H_);

  for (int l = 0; l < L_; ++l) {
    const long long woff = (long long)l * D_ * D_;
    gemm(hb, Wq + woff, bq + l * D_, qb, TP, D_, D_, GF_BIAS);
    gemm(hb, Wk + woff, bk + l * D_, kb, TP, D_, D_, GF_BIAS);
    gemm(hb, Wv + woff, bv + l * D_, vb, TP, D_, D_, GF_BIAS);
    k_attn<<<dim3(P_ * NH_ / 8), blk, 0, stream>>>(qb, kb, vb, atb);
    gemm(atb, Wo + woff, bo + l * D_, tmpb, TP, D_, D_, GF_BIAS);
    k_ln_res<<<dim3((unsigned)((TP + 7) / 8)), blk, 0, stream>>>(hb, tmpb, g1 + l * D_, bg1 + l * D_);
    gemm(hb, Wf1 + (long long)l * D_ * FF_, bf1 + l * FF_, ffb, TP, FF_, D_, GF_BIAS | GF_RELU);
    gemm(ffb, Wf2 + (long long)l * FF_ * D_, bf2 + l * D_, tmpb, TP, D_, FF_, GF_BIAS);
    k_ln_res<<<dim3((unsigned)((TP + 7) / 8)), blk, 0, stream>>>(hb, tmpb, g2 + l * D_, bg2 + l * D_);
  }
  k_rowdot<<<dim3((unsigned)((TP + 7) / 8)), blk, 0, stream>>>(hb, Wout, out + TP * (long long)D_);
}
